// NBodySEGNNWrapper_14843406975348
// MI455X (gfx1250) — matrix operand order repaired and checked by reading, not compiled, not run on hardware
//
#include <hip/hip_runtime.h>
#include <hip/hip_bf16.h>
#include <math.h>

// ---------------- problem constants ----------------
#define NB   1024          // batches
#define NO   128           // objects per batch
#define KNB  32            // neighbors
#define DM   64            // message dim
#define NN   (NB * NO)     // total nodes

static constexpr float Y0 = 0.28209479177387814f;
static constexpr float Y1 = 0.4886025119029199f;

typedef __attribute__((ext_vector_type(16))) _Float16 v16h;
typedef __attribute__((ext_vector_type(8)))  _Float16 v8h;
typedef __attribute__((ext_vector_type(8)))  float    v8f;

// ---------------------------------------------------------------------------
// Kernel 1: per-batch COM, centered positions, node features x=[pos,v,|v|]
// ---------------------------------------------------------------------------
__global__ __launch_bounds__(NO) void k_center(const float* __restrict__ inp,
                                               float* __restrict__ pos,
                                               float* __restrict__ x,
                                               float* __restrict__ com) {
  const int b = blockIdx.x, i = threadIdx.x;
  const float* nd = inp + (size_t)(b * NO + i) * 7;
  const float m  = nd[0];
  const float lx = nd[1], ly = nd[2], lz = nd[3];
  const float vx = nd[4], vy = nd[5], vz = nd[6];

  __shared__ float rm[NO], rx[NO], ry[NO], rz[NO];
  rm[i] = m; rx[i] = m * lx; ry[i] = m * ly; rz[i] = m * lz;
  __syncthreads();
  for (int s = NO / 2; s > 0; s >>= 1) {
    if (i < s) { rm[i] += rm[i+s]; rx[i] += rx[i+s]; ry[i] += ry[i+s]; rz[i] += rz[i+s]; }
    __syncthreads();
  }
  const float inv = 1.0f / rm[0];
  const float cx = rx[0] * inv, cy = ry[0] * inv, cz = rz[0] * inv;

  const int g = b * NO + i;
  const float px = lx - cx, py = ly - cy, pz = lz - cz;
  pos[g*3+0] = px; pos[g*3+1] = py; pos[g*3+2] = pz;
  const float va = sqrtf(vx*vx + vy*vy + vz*vz);
  float* xg = x + (size_t)g * 7;
  xg[0] = px; xg[1] = py; xg[2] = pz;
  xg[3] = vx; xg[4] = vy; xg[5] = vz; xg[6] = va;
  if (i == 0) { com[b*3+0] = cx; com[b*3+1] = cy; com[b*3+2] = cz; }
}

// ---------------------------------------------------------------------------
// Kernel 2: k-NN (top-32 smallest d2, diag excluded, tie -> lowest index,
// matching jax.lax.top_k(-d2)).  One block handles 64 nodes of one batch.
// ---------------------------------------------------------------------------
__global__ __launch_bounds__(64) void k_knn(const float* __restrict__ pos,
                                            int* __restrict__ nbr) {
  const int b = blockIdx.x >> 1;
  const int h = blockIdx.x & 1;
  const int t = threadIdx.x;                  // 0..63

  __shared__ float lp[NO][3];
  __shared__ float row[64][NO];               // 32 KB of d2 rows

  for (int j = t; j < NO; j += 64) {
    lp[j][0] = pos[(size_t)(b*NO + j)*3 + 0];
    lp[j][1] = pos[(size_t)(b*NO + j)*3 + 1];
    lp[j][2] = pos[(size_t)(b*NO + j)*3 + 2];
  }
  __syncthreads();

  const int i = h * 64 + t;
  const float px = lp[i][0], py = lp[i][1], pz = lp[i][2];
  const float BIG = 3.0e38f;
  for (int j = 0; j < NO; ++j) {
    const float dx = px - lp[j][0], dy = py - lp[j][1], dz = pz - lp[j][2];
    row[t][j] = (j == i) ? BIG : (dx*dx + dy*dy + dz*dz);
  }
  // 32x first-occurrence argmin (ties broken by lowest index)
  int* out = nbr + (size_t)(b*NO + i) * KNB;
  for (int k = 0; k < KNB; ++k) {
    float best = BIG; int bj = 0;
    for (int j = 0; j < NO; ++j) {
      const float v = row[t][j];
      if (v < best) { best = v; bj = j; }
    }
    out[k] = b * NO + bj;                     // global flat src index
    row[t][bj] = BIG;
  }
}

// ---------------------------------------------------------------------------
// Kernel 3: fused edge-message GEMM (WMMA f16, K padded 20->32) + ReLU +
// per-dst mean aggregation + sum of unit rel vectors (for node_attr).
// 128 threads = 4 waves; wave w handles dst node g = blockIdx*4 + w and its
// 32 edges (one edge per lane).
// ---------------------------------------------------------------------------
__global__ __launch_bounds__(128) void k_msg(const float* __restrict__ inp,
                                             const float* __restrict__ x,
                                             const int*   __restrict__ nbr,
                                             const float* __restrict__ Wmsg,
                                             float* __restrict__ agg,
                                             float* __restrict__ sum_u) {
  // W_msg staged TRANSPOSED (n-major) so B fragments are contiguous in k.
  __shared__ __align__(16) _Float16 lW[DM][32];      // [n][k], rows 20..31 zero
  __shared__ __align__(16) _Float16 lA[4][32][32];   // per-wave 32x32 m_in tiles

  const int tid  = threadIdx.x;
  const int w    = tid >> 5;
  const int lane = tid & 31;

  // --- stage W_msg (f32 -> f16, zero pad k=20..31) ---
  for (int idx = tid; idx < DM * 32; idx += 128) {
    const int n = idx >> 5, k = idx & 31;
    lW[n][k] = (_Float16)((k < 20) ? Wmsg[k * DM + n] : 0.0f);
  }

  // --- per-lane edge features ---
  const int g   = blockIdx.x * 4 + w;                // dst node (global flat)
  const int src = nbr[(size_t)g * KNB + lane];

  float xs[7], xd[7];
#pragma unroll
  for (int t = 0; t < 7; ++t) { xs[t] = x[(size_t)src*7 + t]; xd[t] = x[(size_t)g*7 + t]; }
  const float rx = xs[0] - xd[0], ry = xs[1] - xd[1], rz = xs[2] - xd[2];
  const float dist = sqrtf(rx*rx + ry*ry + rz*rz);
  const float ir   = 1.0f / fmaxf(dist, 1e-12f);
  const float ux = rx * ir, uy = ry * ir, uz = rz * ir;
  const float pm = inp[(size_t)src*7] * inp[(size_t)g*7];

  float mi[20];
#pragma unroll
  for (int t = 0; t < 7; ++t) { mi[t] = xs[t]; mi[7+t] = xd[t]; }
  mi[14] = Y0; mi[15] = Y1*ux; mi[16] = Y1*uy; mi[17] = Y1*uz;
  mi[18] = dist; mi[19] = pm;

  _Float16* arow = &lA[w][lane][0];
#pragma unroll
  for (int t = 0; t < 20; ++t) arow[t] = (_Float16)mi[t];
#pragma unroll
  for (int t = 20; t < 32; ++t) arow[t] = (_Float16)0.0f;

  __syncthreads();

  // --- load A fragments (16-bit A 16x32 layout: lanes 0-15 K 0..7/16..23,
  //     lanes 16-31 K 8..15/24..31) ---
  const int half = lane >> 4;
  const int lr   = lane & 15;
  const int akb  = half ? 8 : 0;
  const v8h a00 = *(const v8h*)(&lA[w][lr     ][akb]);
  const v8h a01 = *(const v8h*)(&lA[w][lr     ][akb + 16]);
  const v8h a10 = *(const v8h*)(&lA[w][16 + lr][akb]);
  const v8h a11 = *(const v8h*)(&lA[w][16 + lr][akb + 16]);
  v16h A0, A1;
#pragma unroll
  for (int j = 0; j < 8; ++j) {
    A0[j] = a00[j]; A0[8+j] = a01[j];
    A1[j] = a10[j]; A1[8+j] = a11[j];
  }

  // --- 4 column tiles of 16: B 32x16 layout lanes 0-15 K 0..7/16..23, 16-31 K 8..15/24..31 ---
  const int bkb = half ? 8 : 0;
  float sv[4];
#pragma unroll
  for (int ct = 0; ct < 4; ++ct) {
    const _Float16* bcol = &lW[ct*16 + lr][bkb];
    const v8h b0 = *(const v8h*)(bcol);
    const v8h b1 = *(const v8h*)(bcol + 16);
    v16h Bf;
#pragma unroll
    for (int j = 0; j < 8; ++j) { Bf[j] = b0[j]; Bf[8+j] = b1[j]; }

    v8f c0 = {}, c1 = {};
    c0 = __builtin_amdgcn_wmma_f32_16x16x32_f16(false, A0, false, Bf,
                                                (short)0, c0, false, false);
    c1 = __builtin_amdgcn_wmma_f32_16x16x32_f16(false, A1, false, Bf,
                                                (short)0, c1, false, false);
    // ReLU + sum over the 16 M-rows this lane-half owns (both edge tiles)
    float s = 0.0f;
#pragma unroll
    for (int r = 0; r < 8; ++r) s += fmaxf(c0[r], 0.0f) + fmaxf(c1[r], 0.0f);
    // add complementary M half (lane ^ 16) -> full column sum over 32 edges
    s += __shfl_xor(s, 16, 32);
    sv[ct] = s * (1.0f / 32.0f);
  }

  // divergent stores only AFTER all WMMAs (EXEC must be full during WMMA)
  if (half == 0) {
#pragma unroll
    for (int ct = 0; ct < 4; ++ct) agg[(size_t)g * DM + ct*16 + lr] = sv[ct];
  }

  // sum of unit rel vectors over the 32 edges of this dst
  float sx = ux, sy = uy, sz = uz;
#pragma unroll
  for (int off = 16; off > 0; off >>= 1) {
    sx += __shfl_xor(sx, off, 32);
    sy += __shfl_xor(sy, off, 32);
    sz += __shfl_xor(sz, off, 32);
  }
  if (lane == 0) {
    sum_u[(size_t)g*3 + 0] = sx;
    sum_u[(size_t)g*3 + 1] = sy;
    sum_u[(size_t)g*3 + 2] = sz;
  }
}

// ---------------------------------------------------------------------------
// Kernel 4: node_attr + [x, node_attr, agg] @ W_out + loc + com; zero tail.
// ---------------------------------------------------------------------------
__global__ __launch_bounds__(256) void k_out(const float* __restrict__ x,
                                             const float* __restrict__ pos,
                                             const float* __restrict__ com,
                                             const float* __restrict__ sum_u,
                                             const float* __restrict__ agg,
                                             const float* __restrict__ Wout,
                                             float* __restrict__ out) {
  __shared__ float lWo[75 * 3];
  const int tid = threadIdx.x;
  if (tid < 225) lWo[tid] = Wout[tid];
  __syncthreads();

  const int g = blockIdx.x * 256 + tid;     // 0 .. NN-1
  float f[75];
#pragma unroll
  for (int t = 0; t < 7; ++t) f[t] = x[(size_t)g*7 + t];
  const float vx = f[3], vy = f[4], vz = f[5], va = f[6];
  const float iv = 1.0f / fmaxf(va, 1e-12f);
  f[7]  = 2.0f * Y0;
  f[8]  = Y1 * (sum_u[(size_t)g*3+0] * (1.0f/32.0f) + vx * iv);
  f[9]  = Y1 * (sum_u[(size_t)g*3+1] * (1.0f/32.0f) + vy * iv);
  f[10] = Y1 * (sum_u[(size_t)g*3+2] * (1.0f/32.0f) + vz * iv);
#pragma unroll
  for (int j = 0; j < DM; ++j) f[11 + j] = agg[(size_t)g * DM + j];

  const int b = g >> 7;
#pragma unroll
  for (int d = 0; d < 3; ++d) {
    float acc = 0.0f;
#pragma unroll
    for (int j = 0; j < 75; ++j) acc = fmaf(f[j], lWo[j*3 + d], acc);
    out[(size_t)g*3 + d] = pos[(size_t)g*3 + d] + acc + com[b*3 + d];
  }
  if (g < NB) out[(size_t)NN * 3 + g] = 0.0f;  // second tuple output: zeros(B)
}

// ---------------------------------------------------------------------------
// launch
// ---------------------------------------------------------------------------
extern "C" void kernel_launch(void* const* d_in, const int* in_sizes, int n_in,
                              void* d_out, int out_size, void* d_ws, size_t ws_size,
                              hipStream_t stream) {
  const float* inp  = (const float*)d_in[0];   // (1024,128,7)
  const float* Wmsg = (const float*)d_in[1];   // (20,64)
  const float* Wout = (const float*)d_in[2];   // (75,3)
  float* out = (float*)d_out;                  // 1024*128*3 + 1024 floats

  // workspace carve-up (~55 MB total)
  float* pos   = (float*)d_ws;                 // NN*3
  float* x     = pos   + (size_t)NN * 3;       // NN*7
  float* com   = x     + (size_t)NN * 7;       // NB*3
  float* sum_u = com   + (size_t)NB * 3;       // NN*3
  float* agg   = sum_u + (size_t)NN * 3;       // NN*64
  int*   nbr   = (int*)(agg + (size_t)NN * DM);// NN*32 ints

  k_center<<<NB,        NO,  0, stream>>>(inp, pos, x, com);
  k_knn   <<<NB * 2,    64,  0, stream>>>(pos, nbr);
  k_msg   <<<NN / 4,    128, 0, stream>>>(inp, x, nbr, Wmsg, agg, sum_u);
  k_out   <<<NN / 256,  256, 0, stream>>>(x, pos, com, sum_u, agg, Wout, out);
}
